// co_ad_4337916969243
// MI455X (gfx1250) — matrix operand order repaired and checked by reading, not compiled, not run on hardware
//
#include <hip/hip_runtime.h>

// ---------------------------------------------------------------------------
// CDNA5 (gfx1250, wave32) fused contrastive-loss kernel set.
// D = A x B via v_wmma_f32_16x16x32_f16; A rows staged in LDS (padded).
// ---------------------------------------------------------------------------

typedef __attribute__((ext_vector_type(16))) _Float16 v16h;
typedef __attribute__((ext_vector_type(8)))  _Float16 v8h;
typedef __attribute__((ext_vector_type(8)))  float    v8f;

#define BN 4096
#define DK 768
#define ASTRIDE (DK + 8)   // pad 16B per row: 388 dwords % 64 banks = 4 -> no conflicts
#define TINV 20.0f         // 1/TEMP
#define EPSV 1e-6f

// ------------------------- kernel 1: normalize to f16 ----------------------
__global__ void norm_f16_kernel(const float* __restrict__ feat,
                                const float* __restrict__ fgen,
                                const float* __restrict__ faug,
                                _Float16* __restrict__ onh,
                                _Float16* __restrict__ gnh,
                                _Float16* __restrict__ anh,
                                float* __restrict__ out2)
{
    int row   = blockIdx.x;          // 0 .. 3*4096-1
    int which = row / BN;
    int r     = row - which * BN;
    const float* src = (which == 0) ? feat : (which == 1) ? fgen : faug;
    _Float16*    dst = (which == 0) ? onh  : (which == 1) ? gnh  : anh;
    src += (size_t)r * DK;
    dst += (size_t)r * DK;

    float x0 = src[threadIdx.x];
    float x1 = src[threadIdx.x + 256];
    float x2 = src[threadIdx.x + 512];
    float ss = x0 * x0 + x1 * x1 + x2 * x2;
    #pragma unroll
    for (int off = 16; off >= 1; off >>= 1) ss += __shfl_xor(ss, off, 32);

    __shared__ float wsum[8];
    int lane = threadIdx.x & 31, wv = threadIdx.x >> 5;
    if (lane == 0) wsum[wv] = ss;
    __syncthreads();
    float tot = 0.f;
    #pragma unroll
    for (int i = 0; i < 8; ++i) tot += wsum[i];
    float inv = rsqrtf(tot);

    dst[threadIdx.x]       = (_Float16)(x0 * inv);
    dst[threadIdx.x + 256] = (_Float16)(x1 * inv);
    dst[threadIdx.x + 512] = (_Float16)(x2 * inv);

    if (row == 0 && threadIdx.x == 0) { out2[0] = 0.f; out2[1] = 0.f; }
}

// --------------------- WMMA fragment loaders (wave32) ----------------------
// A 16x32 f16: lane L -> row (L&15), 16 halves at kk + 16*(L>=16).
__device__ __forceinline__ v16h load_a_frag(const _Float16* __restrict__ As,
                                            int lane, int kk)
{
    const _Float16* p = As + (lane & 15) * ASTRIDE + kk + ((lane >> 4) << 4);
    v8h lo = *(const v8h*)(p);
    v8h hi = *(const v8h*)(p + 8);
    return __builtin_shufflevector(lo, hi, 0,1,2,3,4,5,6,7,8,9,10,11,12,13,14,15);
}
// B 32x16 f16: lane L -> column row (n0 + (L&15)), 16 halves at kk + 16*(L>=16).
// 32-bit element offset so the backend emits saddr-form global loads.
__device__ __forceinline__ v16h load_b_frag(const _Float16* __restrict__ base,
                                            int boff, int kk)
{
    return *(const v16h*)(base + (boff + kk));
}

// ------------------------- kernel 2: pass 1 (row sums) ---------------------
// sums layout: [4][4096] = { P = s_o+s_a, Q = s_os+s_as, R = s_g, S = s_gs }
__global__ void pass1_kernel(const _Float16* __restrict__ onh,
                             const _Float16* __restrict__ gnh,
                             const _Float16* __restrict__ anh,
                             const int* __restrict__ targets,
                             float* __restrict__ sums)
{
    __shared__ __align__(32) _Float16 As[16 * ASTRIDE];
    __shared__ int   tcol[BN];                        // 16 KB all targets
    __shared__ float red[4][16];

    const int tid  = threadIdx.x;
    const int lane = tid & 31;
    const int wave = tid >> 5;
    const int m0   = blockIdx.x * 16;

    {   // stage A (16 rows, padded) into LDS
        const uint32_t* s = (const uint32_t*)(onh + (size_t)m0 * DK);
        uint32_t*       d = (uint32_t*)As;
        for (int i = tid; i < 16 * DK / 2; i += 256) {
            int rr = i / (DK / 2), cc = i - rr * (DK / 2);
            d[rr * (ASTRIDE / 2) + cc] = s[i];
        }
        for (int i = tid; i < BN; i += 256) tcol[i] = targets[i];
        if (tid < 64) ((float*)red)[tid] = 0.f;
    }
    __syncthreads();

    const int mbase = (lane >> 4) * 8;
    int tm[8];
    #pragma unroll
    for (int v = 0; v < 8; ++v) tm[v] = tcol[m0 + mbase + v];

    float P[8] = {0}, Q[8] = {0}, R[8] = {0}, S[8] = {0};

    for (int nt = wave; nt < BN / 16; nt += 8) {
        const int n0   = nt * 16;
        const int boff = (n0 + (lane & 15)) * DK + ((lane >> 4) << 4);
        __builtin_prefetch(onh + boff, 0, 3);
        __builtin_prefetch(gnh + boff, 0, 3);
        __builtin_prefetch(anh + boff, 0, 3);
        v8f ao = {}, ag = {}, aa = {};
        #pragma unroll 2
        for (int kk = 0; kk < DK; kk += 32) {
            v16h af = load_a_frag(As, lane, kk);
            v16h bo = load_b_frag(onh, boff, kk);
            v16h bg = load_b_frag(gnh, boff, kk);
            v16h ba = load_b_frag(anh, boff, kk);
            ao = __builtin_amdgcn_wmma_f32_16x16x32_f16(false, af, false, bo, (short)0, ao, false, false);
            ag = __builtin_amdgcn_wmma_f32_16x16x32_f16(false, af, false, bg, (short)0, ag, false, false);
            aa = __builtin_amdgcn_wmma_f32_16x16x32_f16(false, af, false, ba, (short)0, aa, false, false);
        }
        const int tn = tcol[n0 + (lane & 15)];
        #pragma unroll
        for (int v = 0; v < 8; ++v) {
            float eo = __expf(ao[v] * TINV);
            float eg = __expf(ag[v] * TINV);
            float ea = __expf(aa[v] * TINV);
            float oa = eo + ea;
            P[v] += oa; R[v] += eg;
            if (tn == tm[v]) { Q[v] += oa; S[v] += eg; }
        }
    }

    // reduce over the 16 lanes sharing each (half, row) pair
    #pragma unroll
    for (int v = 0; v < 8; ++v) {
        #pragma unroll
        for (int off = 1; off < 16; off <<= 1) {
            P[v] += __shfl_xor(P[v], off, 32);
            Q[v] += __shfl_xor(Q[v], off, 32);
            R[v] += __shfl_xor(R[v], off, 32);
            S[v] += __shfl_xor(S[v], off, 32);
        }
    }
    if ((lane & 15) == 0) {
        #pragma unroll
        for (int v = 0; v < 8; ++v) {
            int m = mbase + v;
            atomicAdd(&red[0][m], P[v]);
            atomicAdd(&red[1][m], Q[v]);
            atomicAdd(&red[2][m], R[v]);
            atomicAdd(&red[3][m], S[v]);
        }
    }
    __syncthreads();
    if (tid < 64) {
        int arr = tid >> 4, m = tid & 15;
        sums[(size_t)arr * BN + m0 + m] = red[arr][m];
    }
}

// ------------------------- kernel 3: pass 2 (losses) -----------------------
__global__ void pass2_kernel(const _Float16* __restrict__ onh,
                             const _Float16* __restrict__ gnh,
                             const _Float16* __restrict__ anh,
                             const int* __restrict__ targets,
                             const float* __restrict__ sums,
                             float* __restrict__ out2)   // [ad_loss, co_loss]
{
    __shared__ __align__(32) _Float16 As[16 * ASTRIDE];
    __shared__ int   tcol[BN];
    __shared__ float dco[16], dad[16];
    __shared__ float lred[2][8];

    const int tid  = threadIdx.x;
    const int lane = tid & 31;
    const int wave = tid >> 5;
    const int m0   = blockIdx.x * 16;

    {
        const uint32_t* s = (const uint32_t*)(onh + (size_t)m0 * DK);
        uint32_t*       d = (uint32_t*)As;
        for (int i = tid; i < 16 * DK / 2; i += 256) {
            int rr = i / (DK / 2), cc = i - rr * (DK / 2);
            d[rr * (ASTRIDE / 2) + cc] = s[i];
        }
        for (int i = tid; i < BN; i += 256) tcol[i] = targets[i];
        if (tid < 16) {
            int m = m0 + tid;
            float Pm = sums[0 * BN + m], Qm = sums[1 * BN + m];
            float Rm = sums[2 * BN + m], Sm = sums[3 * BN + m];
            dco[tid] = (Pm - Qm) + Rm + EPSV;
            dad[tid] = (Rm - Sm) + Pm + EPSV;
        }
    }
    __syncthreads();

    const int mbase = (lane >> 4) * 8;
    int   tm[8];
    float dc[8], da[8];
    #pragma unroll
    for (int v = 0; v < 8; ++v) {
        tm[v] = tcol[m0 + mbase + v];
        dc[v] = dco[mbase + v];
        da[v] = dad[mbase + v];
    }

    float ad_acc = 0.f, co_acc = 0.f;

    for (int nt = wave; nt < BN / 16; nt += 8) {
        const int n0   = nt * 16;
        const int boff = (n0 + (lane & 15)) * DK + ((lane >> 4) << 4);
        __builtin_prefetch(onh + boff, 0, 3);
        __builtin_prefetch(gnh + boff, 0, 3);
        __builtin_prefetch(anh + boff, 0, 3);
        v8f ao = {}, ag = {}, aa = {};
        #pragma unroll 2
        for (int kk = 0; kk < DK; kk += 32) {
            v16h af = load_a_frag(As, lane, kk);
            v16h bo = load_b_frag(onh, boff, kk);
            v16h bg = load_b_frag(gnh, boff, kk);
            v16h ba = load_b_frag(anh, boff, kk);
            ao = __builtin_amdgcn_wmma_f32_16x16x32_f16(false, af, false, bo, (short)0, ao, false, false);
            ag = __builtin_amdgcn_wmma_f32_16x16x32_f16(false, af, false, bg, (short)0, ag, false, false);
            aa = __builtin_amdgcn_wmma_f32_16x16x32_f16(false, af, false, ba, (short)0, aa, false, false);
        }
        const int tn = tcol[n0 + (lane & 15)];
        #pragma unroll
        for (int v = 0; v < 8; ++v) {
            if (tn == tm[v]) {
                float eo = __expf(ao[v] * TINV);
                float eg = __expf(ag[v] * TINV);
                float ea = __expf(aa[v] * TINV);
                co_acc -= __logf(eo / (eo + dc[v]) + EPSV) +
                          __logf(ea / (ea + dc[v]) + EPSV);
                ad_acc -= __logf(eg / (eg + da[v]) + EPSV);
            }
        }
    }

    #pragma unroll
    for (int off = 16; off >= 1; off >>= 1) {
        ad_acc += __shfl_xor(ad_acc, off, 32);
        co_acc += __shfl_xor(co_acc, off, 32);
    }
    if (lane == 0) { lred[0][wave] = ad_acc; lred[1][wave] = co_acc; }
    __syncthreads();
    if (tid == 0) {
        float a = 0.f, c = 0.f;
        #pragma unroll
        for (int i = 0; i < 8; ++i) { a += lred[0][i]; c += lred[1][i]; }
        atomicAdd(&out2[0], a * (1.0f / BN));   // ad_loss
        atomicAdd(&out2[1], c * (1.0f / BN));   // co_loss
    }
}

// ---------------------------------------------------------------------------
extern "C" void kernel_launch(void* const* d_in, const int* in_sizes, int n_in,
                              void* d_out, int out_size, void* d_ws, size_t ws_size,
                              hipStream_t stream)
{
    (void)in_sizes; (void)n_in; (void)out_size; (void)ws_size;
    const float* feat    = (const float*)d_in[0];
    const float* fgen    = (const float*)d_in[1];
    const float* faug    = (const float*)d_in[2];
    const int*   targets = (const int*)d_in[3];
    float*       out     = (float*)d_out;

    _Float16* onh  = (_Float16*)d_ws;
    _Float16* gnh  = onh + (size_t)BN * DK;
    _Float16* anh  = gnh + (size_t)BN * DK;
    float*    sums = (float*)(anh + (size_t)BN * DK);   // 4 * 4096 floats

    hipLaunchKernelGGL(norm_f16_kernel, dim3(3 * BN), dim3(256), 0, stream,
                       feat, fgen, faug, onh, gnh, anh, out);
    hipLaunchKernelGGL(pass1_kernel, dim3(BN / 16), dim3(256), 0, stream,
                       onh, gnh, anh, targets, sums);
    hipLaunchKernelGGL(pass2_kernel, dim3(BN / 16), dim3(256), 0, stream,
                       onh, gnh, anh, targets, sums, out);
}
